// Attention_58695023067401
// MI455X (gfx1250) — compile-verified, matrix operand order checked by reading, not run on hardware
//
#include <hip/hip_runtime.h>
#include <hip/hip_bf16.h>

typedef __attribute__((ext_vector_type(16))) __bf16 v16bf;
typedef __attribute__((ext_vector_type(8)))  __bf16 v8bf;
typedef __attribute__((ext_vector_type(8)))  float  v8f;
typedef __attribute__((ext_vector_type(4)))  float  v4f;

namespace {
constexpr int    kB        = 16;
constexpr int    kL        = 1024;
constexpr int    kD        = 256;
constexpr size_t kRows     = (size_t)kB * kL;              // rows per matrix
constexpr size_t kElems    = kRows * (size_t)kD;           // elements per matrix
constexpr size_t kHalfB    = kElems * sizeof(__bf16);      // one bf16 plane: 8 MB
constexpr size_t kNormB    = kRows * sizeof(float);        // 64 KB
constexpr size_t kWsNeeded = 4 * kHalfB + 2 * kNormB;      // ~33.7 MB
}  // namespace

#define WMMA_BF16(A, Bm, C)                                                   \
  __builtin_amdgcn_wmma_f32_16x16x32_bf16(false, (A), false, (Bm), (short)0,  \
                                          (C), false, false)

// ---------------------------------------------------------------------------
// Pass 1: split fp32 -> bf16 hi/lo planes + exact fp32 row norms.
// One wave per row of 256 floats (8 floats/lane). Streaming, memory-bound.
// Source is read-once -> nontemporal loads; planes stay resident in L2.
// ---------------------------------------------------------------------------
__global__ __launch_bounds__(256)
void split_norm_kernel(const float* __restrict__ src,
                       __bf16* __restrict__ hi, __bf16* __restrict__ lo,
                       float* __restrict__ norms) {
  const int lane = threadIdx.x & 31;
  const int row  = blockIdx.x * 8 + (threadIdx.x >> 5);
  const size_t off = (size_t)row * kD + lane * 8;

  v4f x0 = __builtin_nontemporal_load((const v4f*)(src + off));
  v4f x1 = __builtin_nontemporal_load((const v4f*)(src + off + 4));
  v8bf h, l;
  float nrm = 0.0f;
#pragma unroll
  for (int t = 0; t < 4; ++t) {
    float f0 = x0[t], f1 = x1[t];
    nrm = fmaf(f0, f0, nrm);
    nrm = fmaf(f1, f1, nrm);
    __bf16 h0 = (__bf16)f0, h1 = (__bf16)f1;
    h[t] = h0;  h[4 + t] = h1;
    l[t]     = (__bf16)(f0 - (float)h0);
    l[4 + t] = (__bf16)(f1 - (float)h1);
  }
  *(v8bf*)(hi + off) = h;
  *(v8bf*)(lo + off) = l;

#pragma unroll
  for (int s = 16; s > 0; s >>= 1) nrm += __shfl_xor(nrm, s, 32);
  if (lane == 0) norms[row] = nrm;
}

// ---------------------------------------------------------------------------
// Load one 16x32 bf16 WMMA fragment in ISA lane layout:
// elems 0..7 = K[p..p+7], elems 8..15 = K[p+16..p+23] (p already includes the
// per-half-wave +8 shift). Two 16-byte loads.
// ---------------------------------------------------------------------------
__device__ inline v16bf load_frag(const __bf16* __restrict__ p) {
  v8bf a = *(const v8bf*)p;
  v8bf b = *(const v8bf*)(p + 16);
  return __builtin_shufflevector(a, b, 0, 1, 2, 3, 4, 5, 6, 7,
                                 8, 9, 10, 11, 12, 13, 14, 15);
}

// ---------------------------------------------------------------------------
// Pass 2: pure-WMMA GEMM + fused epilogue. Each wave owns a 32x64 output
// tile (2x4 accumulators -> 24 WMMA per 24 b128 loads per K-chunk);
// block = 8 waves (2x4) covering 64x256 of output.
// ---------------------------------------------------------------------------
__global__ __launch_bounds__(256)
void pdist_wmma_gemm_kernel(const __bf16* __restrict__ s1h,
                            const __bf16* __restrict__ s1l,
                            const __bf16* __restrict__ s2h,
                            const __bf16* __restrict__ s2l,
                            const float* __restrict__ n1,
                            const float* __restrict__ n2,
                            float* __restrict__ out) {
  const int lane  = threadIdx.x & 31;
  const int wave  = threadIdx.x >> 5;
  const int batch = blockIdx.z;
  const int i0 = blockIdx.y * 64  + (wave >> 2) * 32;   // 32 output rows
  const int j0 = blockIdx.x * 256 + (wave & 3)  * 64;   // 64 output cols
  const int m     = lane & 15;
  const int khalf = (lane >> 4) * 8;

  const size_t rb = (size_t)batch * kL;  // batch row base
  const __bf16* a_h = s1h + (rb + i0 + m) * kD + khalf;
  const __bf16* a_l = s1l + (rb + i0 + m) * kD + khalf;
  const __bf16* b_h = s2h + (rb + j0 + m) * kD + khalf;
  const __bf16* b_l = s2l + (rb + j0 + m) * kD + khalf;
  constexpr int kRow16 = 16 * kD;  // 16-row fragment stride (elements)

  v8f c[2][4];
#pragma unroll
  for (int ti = 0; ti < 2; ++ti)
#pragma unroll
    for (int tj = 0; tj < 4; ++tj) c[ti][tj] = (v8f){};

#pragma unroll 2
  for (int kc = 0; kc < kD; kc += 32) {
    v16bf Ah[2], Al[2], Bh[4], Bl[4];
#pragma unroll
    for (int ti = 0; ti < 2; ++ti) {
      Ah[ti] = load_frag(a_h + ti * kRow16 + kc);
      Al[ti] = load_frag(a_l + ti * kRow16 + kc);
    }
#pragma unroll
    for (int tj = 0; tj < 4; ++tj) {
      Bh[tj] = load_frag(b_h + tj * kRow16 + kc);
      Bl[tj] = load_frag(b_l + tj * kRow16 + kc);
    }

    // bf16x3 = hi*lo + lo*hi + hi*hi. Grouped by term so same-accumulator
    // WMMAs are 8 issues apart (>= RAW hazard distance).
#pragma unroll
    for (int ti = 0; ti < 2; ++ti)
#pragma unroll
      for (int tj = 0; tj < 4; ++tj)
        c[ti][tj] = WMMA_BF16(Ah[ti], Bl[tj], c[ti][tj]);
#pragma unroll
    for (int ti = 0; ti < 2; ++ti)
#pragma unroll
      for (int tj = 0; tj < 4; ++tj)
        c[ti][tj] = WMMA_BF16(Al[ti], Bh[tj], c[ti][tj]);
#pragma unroll
    for (int ti = 0; ti < 2; ++ti)
#pragma unroll
      for (int tj = 0; tj < 4; ++tj)
        c[ti][tj] = WMMA_BF16(Ah[ti], Bh[tj], c[ti][tj]);
  }

  // Epilogue: out = 1 / (1 + sqrt(max(x2 + y2 - 2*dot, 0))).
  // C/D layout: VGPR r, lane l -> M = r + 8*(l>=16), N = l%16.
  const int rbase = (lane >> 4) * 8;
  const float* n1b = n1 + rb;
  const float* n2b = n2 + rb;
  float y2[4];
#pragma unroll
  for (int tj = 0; tj < 4; ++tj) y2[tj] = n2b[j0 + tj * 16 + m];

  float* ob = out + ((size_t)batch * kL + i0) * kL + j0 + m;
#pragma unroll
  for (int ti = 0; ti < 2; ++ti) {
#pragma unroll
    for (int r = 0; r < 8; ++r) {
      const int row = ti * 16 + rbase + r;
      const float x2 = n1b[i0 + row];
      float* orow = ob + (size_t)row * kL;
#pragma unroll
      for (int tj = 0; tj < 4; ++tj) {
        float sq = fmaxf(x2 + y2[tj] - 2.0f * c[ti][tj][r], 0.0f);
        // Output is write-once: nontemporal keeps the bf16 planes in L2.
        __builtin_nontemporal_store(1.0f / (1.0f + sqrtf(sq)),
                                    orow + tj * 16);
      }
    }
  }
}

// ---------------------------------------------------------------------------
// Fallback (small workspace): fused kernel — split in registers.
// ---------------------------------------------------------------------------
namespace {
__device__ inline void split4(v4f x, v16bf& hi, v16bf& lo, int base,
                              float& nrm) {
#pragma unroll
  for (int t = 0; t < 4; ++t) {
    float f = x[t];
    nrm = fmaf(f, f, nrm);
    __bf16 h = (__bf16)f;
    hi[base + t] = h;
    lo[base + t] = (__bf16)(f - (float)h);
  }
}
}  // namespace

__global__ __launch_bounds__(256)
void pdist_fused_fallback_kernel(const float* __restrict__ s1,
                                 const float* __restrict__ s2,
                                 float* __restrict__ out) {
  const int lane  = threadIdx.x & 31;
  const int wave  = threadIdx.x >> 5;
  const int batch = blockIdx.z;
  const int i0 = blockIdx.y * 32 + (wave >> 2) * 16;
  const int j0 = blockIdx.x * 64 + (wave & 3) * 16;
  const int m     = lane & 15;
  const int khalf = (lane >> 4) * 8;

  const float* arow = s1 + ((size_t)batch * kL + (size_t)(i0 + m)) * kD;
  const float* brow = s2 + ((size_t)batch * kL + (size_t)(j0 + m)) * kD;

  v8f acc = {};
  float an = 0.0f, bn = 0.0f;
  for (int kc = 0; kc < kD; kc += 32) {
    const float* ap = arow + kc + khalf;
    const float* bp = brow + kc + khalf;
    v16bf ahi, alo, bhi, blo;
    split4(*(const v4f*)(ap + 0),  ahi, alo, 0,  an);
    split4(*(const v4f*)(ap + 4),  ahi, alo, 4,  an);
    split4(*(const v4f*)(ap + 16), ahi, alo, 8,  an);
    split4(*(const v4f*)(ap + 20), ahi, alo, 12, an);
    split4(*(const v4f*)(bp + 0),  bhi, blo, 0,  bn);
    split4(*(const v4f*)(bp + 4),  bhi, blo, 4,  bn);
    split4(*(const v4f*)(bp + 16), bhi, blo, 8,  bn);
    split4(*(const v4f*)(bp + 20), bhi, blo, 12, bn);
    acc = WMMA_BF16(ahi, blo, acc);
    acc = WMMA_BF16(alo, bhi, acc);
    acc = WMMA_BF16(ahi, bhi, acc);
  }
  an += __shfl_xor(an, 16, 32);
  bn += __shfl_xor(bn, 16, 32);
  const int rbase = (lane >> 4) * 8;
  float* obase = out + (size_t)batch * kL * kL + (size_t)i0 * kL + j0 + m;
#pragma unroll
  for (int r = 0; r < 8; ++r) {
    float rn = __shfl(an, rbase + r, 32);
    float sq = fmaxf(rn + bn - 2.0f * acc[r], 0.0f);
    obase[(size_t)(rbase + r) * kL] = 1.0f / (1.0f + sqrtf(sq));
  }
}

// ---------------------------------------------------------------------------
extern "C" void kernel_launch(void* const* d_in, const int* in_sizes, int n_in,
                              void* d_out, int out_size, void* d_ws,
                              size_t ws_size, hipStream_t stream) {
  const float* s1 = (const float*)d_in[1];
  const float* s2 = (const float*)d_in[2];
  float* out = (float*)d_out;

  if (ws_size >= kWsNeeded) {
    char* ws = (char*)d_ws;
    __bf16* s1h = (__bf16*)(ws);
    __bf16* s1l = (__bf16*)(ws + kHalfB);
    __bf16* s2h = (__bf16*)(ws + 2 * kHalfB);
    __bf16* s2l = (__bf16*)(ws + 3 * kHalfB);
    float*  n1  = (float*)(ws + 4 * kHalfB);
    float*  n2  = (float*)(ws + 4 * kHalfB + kNormB);

    dim3 cg((unsigned)(kRows / 8));
    split_norm_kernel<<<cg, 256, 0, stream>>>(s1, s1h, s1l, n1);
    split_norm_kernel<<<cg, 256, 0, stream>>>(s2, s2h, s2l, n2);

    dim3 grid(kL / 256, kL / 64, kB);
    pdist_wmma_gemm_kernel<<<grid, 256, 0, stream>>>(s1h, s1l, s2h, s2l, n1,
                                                     n2, out);
  } else {
    dim3 grid(kL / 64, kL / 32, kB);
    pdist_fused_fallback_kernel<<<grid, 256, 0, stream>>>(s1, s2, out);
  }
}
